// VolumeRenderer_88828513616631
// MI455X (gfx1250) — matrix operand order repaired and checked by reading, not compiled, not run on hardware
//
#include <hip/hip_runtime.h>
#include <hip/hip_bf16.h>

typedef __attribute__((ext_vector_type(16))) _Float16 v16h;
typedef __attribute__((ext_vector_type(8)))  _Float16 v8h;
typedef __attribute__((ext_vector_type(8)))  float    v8f;

#define S_DIM   4
#define R_DIM   2048
#define K_STEPS 128
#define C_CH    32
#define RES     128
#define GRID_N  128
#define HID     128

// packed weight table sizes (halves)
#define W1_HALFS (3*8*32*16)   // 12288
#define W2_HALFS (4*8*32*16)   // 16384
#define W3_HALFS (4*1*32*16)   // 2048

// ---------------------------------------------------------------------------
// Pre-pack W1/W2/W3 (f32, row-major [K][N]) into WMMA B-fragment layout, f16.
// B 32x16 (16-bit): lane L holds column n=L&15; halves j=0..15 hold
// K = 8*(L>>4) + (j&7) + 16*(j>>3).  Fragment = [kk][t][lane][16].
// ---------------------------------------------------------------------------
__global__ __launch_bounds__(256) void prepack_weights(
    const float* __restrict__ W1, const float* __restrict__ W2,
    const float* __restrict__ W3, _Float16* __restrict__ ws)
{
  int tid = threadIdx.x;
  for (int e = tid; e < W1_HALFS; e += 256) {
    int j = e & 15, lane = (e >> 4) & 31, t = (e >> 9) & 7, kk = e >> 12;
    int n = lane & 15, kb = (lane < 16) ? 0 : 16;
    int k = 32 * kk + (kb >> 1) + (j & 7) + ((j >> 3) << 4);
    ws[e] = (_Float16)W1[k * HID + 16 * t + n];
  }
  for (int e = tid; e < W2_HALFS; e += 256) {
    int j = e & 15, lane = (e >> 4) & 31, t = (e >> 9) & 7, kk = e >> 12;
    int n = lane & 15, kb = (lane < 16) ? 0 : 16;
    int k = 32 * kk + (kb >> 1) + (j & 7) + ((j >> 3) << 4);
    ws[W1_HALFS + e] = (_Float16)W2[k * HID + 16 * t + n];
  }
  for (int e = tid; e < W3_HALFS; e += 256) {
    int j = e & 15, lane = (e >> 4) & 31, kk = e >> 9;
    int n = lane & 15, kb = (lane < 16) ? 0 : 16;
    int k = 32 * kk + (kb >> 1) + (j & 7) + ((j >> 3) << 4);
    ws[W1_HALFS + W2_HALFS + e] =
        (n < 4) ? (_Float16)W3[k * 4 + n] : (_Float16)0.0f;
  }
}

// ---------------------------------------------------------------------------
// Main renderer: one wave == one ray; 8 M-tiles of 16 samples each.
// ---------------------------------------------------------------------------
__global__ __launch_bounds__(256) void nerf_render(
    const float* __restrict__ rays_o, const float* __restrict__ rays_d,
    const float* __restrict__ code,   const int* __restrict__ dgrid,
    const float* __restrict__ b1, const float* __restrict__ b2,
    const float* __restrict__ b3, const _Float16* __restrict__ wsW,
    float* __restrict__ out_wsum, float* __restrict__ out_depth,
    float* __restrict__ out_img)
{
  __shared__ __attribute__((aligned(16))) _Float16 sH[8][16 * 136]; // per-wave h staging
  __shared__ float sStage[8][64];                                    // per-wave sigma/rgb

  const int lane = threadIdx.x & 31;
  const int wv   = threadIdx.x >> 5;
  const int ray  = blockIdx.x * 8 + wv;     // grid sized so ray < S*R exactly
  const int s    = ray / R_DIM;

  const int m     = lane & 15;              // A row / C column index
  const int hlf   = lane >> 4;              // half-wave select
  const int n     = m;                      // C/D column for this lane
  const int cbase = hlf * 8;                // A-fragment K base within slice

  // per-lane bias registers (column n of each 16-wide N tile)
  float rb1[8], rb2[8];
#pragma unroll
  for (int t = 0; t < 8; ++t) { rb1[t] = b1[t * 16 + n]; rb2[t] = b2[t * 16 + n]; }
  const float rb3 = (n < 4) ? b3[n] : 0.0f;

  // ray setup (identical across all lanes of the wave)
  const float ox = rays_o[ray * 3 + 0], oy = rays_o[ray * 3 + 1], oz = rays_o[ray * 3 + 2];
  const float dx = rays_d[ray * 3 + 0], dy = rays_d[ray * 3 + 1], dz = rays_d[ray * 3 + 2];
  const float sdx = (fabsf(dx) < 1e-9f) ? 1e-9f : dx;
  const float sdy = (fabsf(dy) < 1e-9f) ? 1e-9f : dy;
  const float sdz = (fabsf(dz) < 1e-9f) ? 1e-9f : dz;
  const float t1x = (-1.f - ox) / sdx, t2x = (1.f - ox) / sdx;
  const float t1y = (-1.f - oy) / sdy, t2y = (1.f - oy) / sdy;
  const float t1z = (-1.f - oz) / sdz, t2z = (1.f - oz) / sdz;
  float nearT = fmaxf(fmaxf(fminf(t1x, t2x), fminf(t1y, t2y)), fminf(t1z, t2z));
  float farT  = fminf(fminf(fmaxf(t1x, t2x), fmaxf(t1y, t2y)), fmaxf(t1z, t2z));
  nearT = fmaxf(nearT, 0.2f);
  farT  = fmaxf(farT, nearT);
  const float delta = (farT - nearT) * (1.0f / K_STEPS);

  const float* codeS = code + (size_t)s * 3 * C_CH * RES * RES;

  // Offset that is always 0, but an empty asm per tile iteration makes it
  // opaque to the optimizer: blocks LICM from hoisting the 60 B-fragment
  // loads (which previously blew past 256 VGPRs and spilled), while keeping
  // the loads rooted at the kernarg pointer so address-space inference still
  // produces global_load_b128 (LOADcnt only) instead of flat (LOADcnt+DScnt).
  int wofs = 0;

  float carry = 0.f, wsum = 0.f, dep = 0.f, imr = 0.f, img = 0.f, imb = 0.f;
  const v8f zero8 = {0.f, 0.f, 0.f, 0.f, 0.f, 0.f, 0.f, 0.f};

  for (int tile = 0; tile < 8; ++tile) {
    asm volatile("" : "+s"(wofs));          // break loop-invariance (wofs==0)
    const _Float16* w1p = wsW + wofs;
    const _Float16* w2p = wsW + W1_HALFS + wofs;
    const _Float16* w3p = wsW + W1_HALFS + W2_HALFS + wofs;

    const int ks = tile * 16 + m;
    const float zV = nearT + (farT - nearT) * ((ks + 0.5f) * (1.0f / K_STEPS));
    const float px = fminf(fmaxf(ox + dx * zV, -1.f), 1.f);
    const float py = fminf(fmaxf(oy + dy * zV, -1.f), 1.f);
    const float pz = fminf(fmaxf(oz + dz * zV, -1.f), 1.f);

    // ---------------- layer 1: feats(96) @ W1 -> 16x128 -------------------
    v8f acc[8];
#pragma unroll
    for (int t = 0; t < 8; ++t) acc[t] = zero8;

#pragma unroll
    for (int p = 0; p < 3; ++p) {           // K-slice == plane
      const float u = (p == 2) ? py : px;
      const float v = (p == 0) ? py : pz;
      const float fx = (u + 1.f) * 0.5f * (RES - 1);
      const float fy = (v + 1.f) * 0.5f * (RES - 1);
      int x0 = (int)floorf(fx); x0 = (x0 < 0) ? 0 : ((x0 > RES - 2) ? RES - 2 : x0);
      int y0 = (int)floorf(fy); y0 = (y0 < 0) ? 0 : ((y0 > RES - 2) ? RES - 2 : y0);
      const float wx = fx - (float)x0, wy = fy - (float)y0;
      const float w00 = (1.f - wx) * (1.f - wy), w01 = wx * (1.f - wy);
      const float w10 = (1.f - wx) * wy,         w11 = wx * wy;
      const float* pb = codeS + (size_t)p * C_CH * RES * RES + (size_t)y0 * RES + x0;

      v16h A;
#pragma unroll
      for (int j = 0; j < 16; ++j) {
        const int c = cbase + (j & 7) + ((j >> 3) << 4);   // ISA A-layout channel
        const float* pc = pb + (size_t)c * RES * RES;
        const float f00 = pc[0], f01 = pc[1], f10 = pc[RES], f11 = pc[RES + 1];
        A[j] = (_Float16)(f00 * w00 + f01 * w01 + f10 * w10 + f11 * w11);
      }
#pragma unroll
      for (int t = 0; t < 8; ++t) {
        v16h B = *(const v16h*)(w1p + ((size_t)((p * 8 + t) * 32 + lane)) * 16);
        acc[t] = __builtin_amdgcn_wmma_f32_16x16x32_f16(
            false, A, false, B, (short)0, acc[t], false, false);
      }
    }

    // bias + ReLU, stage h (row-major, stride 136 halves => 16B aligned rows)
#pragma unroll
    for (int t = 0; t < 8; ++t) {
#pragma unroll
      for (int v8 = 0; v8 < 8; ++v8) {
        const float h = fmaxf(acc[t][v8] + rb1[t], 0.f);
        sH[wv][(v8 + 8 * hlf) * 136 + t * 16 + n] = (_Float16)h;
      }
    }
    __syncthreads();

    // ---------------- layer 2: h(128) @ W2 -> 16x128 ----------------------
    v8f acc2[8];
#pragma unroll
    for (int t = 0; t < 8; ++t) acc2[t] = zero8;
#pragma unroll
    for (int kk = 0; kk < 4; ++kk) {
      const _Float16* hp = &sH[wv][m * 136 + 32 * kk + cbase];
      v8h lo = *(const v8h*)hp;
      v8h hi = *(const v8h*)(hp + 16);
      v16h A2 = __builtin_shufflevector(lo, hi, 0, 1, 2, 3, 4, 5, 6, 7,
                                        8, 9, 10, 11, 12, 13, 14, 15);
#pragma unroll
      for (int t = 0; t < 8; ++t) {
        v16h B = *(const v16h*)(w2p + ((size_t)((kk * 8 + t) * 32 + lane)) * 16);
        acc2[t] = __builtin_amdgcn_wmma_f32_16x16x32_f16(
            false, A2, false, B, (short)0, acc2[t], false, false);
      }
    }
    __syncthreads();

    // bias + ReLU, restage h2
#pragma unroll
    for (int t = 0; t < 8; ++t) {
#pragma unroll
      for (int v8 = 0; v8 < 8; ++v8) {
        const float h = fmaxf(acc2[t][v8] + rb2[t], 0.f);
        sH[wv][(v8 + 8 * hlf) * 136 + t * 16 + n] = (_Float16)h;
      }
    }
    __syncthreads();

    // ---------------- layer 3: h2(128) @ W3 -> 16x4 (padded to 16) --------
    v8f acc3 = zero8;
#pragma unroll
    for (int kk = 0; kk < 4; ++kk) {
      const _Float16* hp = &sH[wv][m * 136 + 32 * kk + cbase];
      v8h lo = *(const v8h*)hp;
      v8h hi = *(const v8h*)(hp + 16);
      v16h A3 = __builtin_shufflevector(lo, hi, 0, 1, 2, 3, 4, 5, 6, 7,
                                        8, 9, 10, 11, 12, 13, 14, 15);
      v16h B = *(const v16h*)(w3p + ((size_t)(kk * 32 + lane)) * 16);
      acc3 = __builtin_amdgcn_wmma_f32_16x16x32_f16(
          false, A3, false, B, (short)0, acc3, false, false);
    }

    // activations -> sStage[sample][4]  (col 0: softplus, cols 1..3: sigmoid)
    if (n < 4) {
#pragma unroll
      for (int v8 = 0; v8 < 8; ++v8) {
        const float x = acc3[v8] + rb3;
        const float val = (n == 0)
            ? ((x > 20.f) ? x : log1pf(expf(x)))
            : (1.f / (1.f + expf(-x)));
        sStage[wv][(v8 + 8 * hlf) * 4 + n] = val;
      }
    }
    __syncthreads();

    // ---------------- compositing: 16-wide shfl scan ----------------------
    {
      float sig = sStage[wv][m * 4 + 0];
      const float cr = sStage[wv][m * 4 + 1];
      const float cg = sStage[wv][m * 4 + 2];
      const float cb = sStage[wv][m * 4 + 3];

      int gx = (int)((px + 1.f) * 0.5f * GRID_N); gx = (gx < 0) ? 0 : ((gx > GRID_N - 1) ? GRID_N - 1 : gx);
      int gy = (int)((py + 1.f) * 0.5f * GRID_N); gy = (gy < 0) ? 0 : ((gy > GRID_N - 1) ? GRID_N - 1 : gy);
      int gz = (int)((pz + 1.f) * 0.5f * GRID_N); gz = (gz < 0) ? 0 : ((gz > GRID_N - 1) ? GRID_N - 1 : gz);
      const int occ = dgrid[(((size_t)s * GRID_N + gx) * GRID_N + gy) * GRID_N + gz];
      if (occ <= 0) sig = 0.f;

      const float tau = sig * delta;
      float xs = tau;                          // inclusive scan over 16 lanes
#pragma unroll
      for (int off = 1; off < 16; off <<= 1) {
        const float ysh = __shfl_up(xs, (unsigned)off, 16);
        if (m >= off) xs += ysh;
      }
      const float excl  = xs - tau;
      const float T     = expf(-(carry + excl));
      const float alpha = 1.f - expf(-tau);
      const float wgt   = (T > 1e-4f) ? alpha * T : 0.f;

      float a0 = wgt, a1 = wgt * zV, a2 = wgt * cr, a3 = wgt * cg, a4 = wgt * cb;
#pragma unroll
      for (int off = 8; off >= 1; off >>= 1) {
        a0 += __shfl_xor(a0, off, 16);
        a1 += __shfl_xor(a1, off, 16);
        a2 += __shfl_xor(a2, off, 16);
        a3 += __shfl_xor(a3, off, 16);
        a4 += __shfl_xor(a4, off, 16);
      }
      wsum += a0; dep += a1; imr += a2; img += a3; imb += a4;
      carry += __shfl(xs, 15, 16);             // total tau of this tile
    }
    __syncthreads();
  }

  if (lane == 0) {
    out_wsum[ray]  = wsum;
    out_depth[ray] = dep;
    out_img[ray * 3 + 0] = imr;
    out_img[ray * 3 + 1] = img;
    out_img[ray * 3 + 2] = imb;
  }
}

// ---------------------------------------------------------------------------
extern "C" void kernel_launch(void* const* d_in, const int* in_sizes, int n_in,
                              void* d_out, int out_size, void* d_ws, size_t ws_size,
                              hipStream_t stream)
{
  const float* rays_o = (const float*)d_in[0];
  const float* rays_d = (const float*)d_in[1];
  const float* code   = (const float*)d_in[2];
  const int*   dgrid  = (const int*)d_in[3];
  // d_in[4] = grid_size scalar (== 128, compiled in)
  const float* W1 = (const float*)d_in[5];
  const float* b1 = (const float*)d_in[6];
  const float* W2 = (const float*)d_in[7];
  const float* b2 = (const float*)d_in[8];
  const float* W3 = (const float*)d_in[9];
  const float* b3 = (const float*)d_in[10];

  float*     out = (float*)d_out;
  _Float16*  ws  = (_Float16*)d_ws;   // needs 61440 bytes

  prepack_weights<<<1, 256, 0, stream>>>(W1, W2, W3, ws);

  const int nrays = S_DIM * R_DIM;           // 8192, one wave per ray
  nerf_render<<<nrays / 8, 256, 0, stream>>>(rays_o, rays_d, code, dgrid,
                                             b1, b2, b3, ws,
                                             out, out + nrays, out + 2 * nrays);
}
